// UniF_Att_CNN_18073222381857
// MI455X (gfx1250) — matrix operand order repaired and checked by reading, not compiled, not run on hardware
//
#include <hip/hip_runtime.h>
#include <hip/hip_bf16.h>
#include <math.h>

// ---------------- problem constants ----------------
#define N_UTTS   2048
#define SEQ_LEN  50
#define D_EMB    300
#define D_FILT   64
#define D_OUT    100
#define D_CAT    192          // 3 * 64
#define WIND     10
#define HOPS     3
#define N_CLASSES 7
#define NEG_INF_REF (-1.0e10f)

// conv K padded to multiples of 32 for the f16 WMMA (16x16x32)
#define KPAD3 928             // 3*300 = 900  -> 928
#define KPAD4 1216            // 4*300 = 1200 -> 1216
#define KPAD5 1504            // 5*300 = 1500 -> 1504
#define WOFF3 0
#define WOFF4 (64*KPAD3)              // 59392
#define WOFF5 (WOFF4 + 64*KPAD4)      // 137216
#define WTOT  (WOFF5 + 64*KPAD5)      // 233472 halves

// GRU WMMA padding: K 100->128, N 300->304
#define GRU_KP 128
#define GRU_NP 304
#define GRU_W  (GRU_NP * GRU_KP)      // 38912 halves per matrix
#define GW     16                     // windows per GRU block

#define PRED_ELEMS (N_UTTS * N_CLASSES)          // 14336
#define ATTN_STRIDE ((N_UTTS - 1) * WIND)        // 20470

typedef __attribute__((ext_vector_type(16))) _Float16 v16h;
typedef __attribute__((ext_vector_type(8)))  _Float16 v8h;
typedef __attribute__((ext_vector_type(4)))  _Float16 v4h;
typedef __attribute__((ext_vector_type(8)))  float    v8f;

union AFrag { v16h v; v4h c[4]; };
union BFrag { v16h v; v8h c[2]; };

// float max via integer atomics (works for mixed signs; init = -inf bits)
__device__ __forceinline__ void atomicMaxFloatBits(int* addr, float v) {
    if (v >= 0.0f) atomicMax(addr, __float_as_int(v));
    else           atomicMin((unsigned int*)addr, __float_as_uint(v));
}

// ---------------------------------------------------------------------------
// Kernel 0a: conv weights (64,1,fs,300) f32 -> (64, Kpad) f16, zero padded
// ---------------------------------------------------------------------------
__global__ void prep_weights_kernel(const float* __restrict__ w3,
                                    const float* __restrict__ w4,
                                    const float* __restrict__ w5,
                                    _Float16* __restrict__ wf16) {
    const int fsIdx = blockIdx.y;            // 0,1,2 -> fs 3,4,5
    const int f     = blockIdx.x;            // filter 0..63
    const int fs    = 3 + fsIdx;
    const int K     = fs * D_EMB;
    const int Kpad  = (fsIdx == 0) ? KPAD3 : (fsIdx == 1) ? KPAD4 : KPAD5;
    const int base  = (fsIdx == 0) ? WOFF3 : (fsIdx == 1) ? WOFF4 : WOFF5;
    const float* w  = (fsIdx == 0) ? w3 : (fsIdx == 1) ? w4 : w5;
    for (int k = threadIdx.x; k < Kpad; k += blockDim.x) {
        float v = (k < K) ? w[f * K + k] : 0.0f;
        wf16[base + f * Kpad + k] = (_Float16)v;
    }
}

// ---------------------------------------------------------------------------
// Kernel 0b: GRU weights (300,100) f32 -> (304,128) f16, zero padded
// ---------------------------------------------------------------------------
__global__ void prep_gru_kernel(const float* __restrict__ wih,
                                const float* __restrict__ whh,
                                _Float16* __restrict__ wihB,
                                _Float16* __restrict__ whhB) {
    int i = blockIdx.x * blockDim.x + threadIdx.x;
    if (i >= GRU_W) return;
    int n = i / GRU_KP, k = i % GRU_KP;
    bool ok = (n < 3 * D_OUT) && (k < D_OUT);
    wihB[i] = (_Float16)(ok ? wih[n * D_OUT + k] : 0.0f);
    whhB[i] = (_Float16)(ok ? whh[n * D_OUT + k] : 0.0f);
}

// ---------------------------------------------------------------------------
// Kernel 1: per-utterance CNN encoder. 12 wave32s, one (fsIdx, ntile) WMMA
// job per wave with 3 M-accumulators (B loaded once per K-step, used 3x).
// ---------------------------------------------------------------------------
__global__ void __launch_bounds__(384)
conv_encoder_kernel(const int*   __restrict__ sents,
                    const float* __restrict__ emb,
                    const float* __restrict__ cb3,
                    const float* __restrict__ cb4,
                    const float* __restrict__ cb5,
                    const float* __restrict__ trans_w,
                    const float* __restrict__ trans_b,
                    const _Float16* __restrict__ wf16,
                    float* __restrict__ s_utt) {
    __shared__ _Float16 E[16384];        // 50x300 f16 + zeroed tail (overread pad)
    __shared__ int   featbits[D_CAT];
    __shared__ float feats[D_CAT];

    const int n   = blockIdx.x;
    const int tid = threadIdx.x;

    // --- stage embeddings into LDS as f16 (float4 granularity, 75 per row) ---
    for (int i = tid; i < SEQ_LEN * (D_EMB / 4); i += blockDim.x) {
        int row = i / (D_EMB / 4);
        int c4  = i % (D_EMB / 4);
        int word = sents[n * SEQ_LEN + row];
        const float4 v = ((const float4*)(emb + (size_t)word * D_EMB))[c4];
        v4h h = { (_Float16)v.x, (_Float16)v.y, (_Float16)v.z, (_Float16)v.w };
        *(v4h*)&E[row * D_EMB + c4 * 4] = h;
    }
    for (int i = SEQ_LEN * D_EMB + tid; i < 16384; i += blockDim.x)
        E[i] = (_Float16)0.0f;
    if (tid < D_CAT) featbits[tid] = __float_as_int(-INFINITY);
    __syncthreads();

    const int wave  = tid >> 5;          // 0..11 -> one tile job per wave
    const int lane  = tid & 31;
    const int laneM = lane & 15;
    const int hs    = lane >> 4;         // which K-half this lane owns

    const int KPAD[3]  = { KPAD3, KPAD4, KPAD5 };
    const int WBASE[3] = { WOFF3, WOFF4, WOFF5 };

    const int fsIdx = wave >> 2;         // 0..2
    const int nt    = wave & 3;          // 0..3
    const int Kpad  = KPAD[fsIdx];
    const int P     = 48 - fsIdx;        // valid positions: 48/47/46
    const int ncol  = nt * 16 + laneM;
    const _Float16* brow = wf16 + WBASE[fsIdx] + ncol * Kpad;

    v8f acc[3] = { {}, {}, {} };
    for (int ks = 0; ks < Kpad; ks += 32) {
        // B fragment: lane%16 = N column; lane<16 -> K {0..7,16..23}; lane>=16 -> K {8..15,24..31}
        const _Float16* bp = brow + ks + hs * 8;
        BFrag bf;
        bf.c[0] = *(const v8h*)(bp);
        bf.c[1] = *(const v8h*)(bp + 16);
        #pragma unroll
        for (int m = 0; m < 3; ++m) {
            // A fragment: lane<16 -> K {0..7,16..23}; lane>=16 -> K {8..15,24..31}
            const int a0 = (m * 16 + laneM) * D_EMB + ks + hs * 8;
            AFrag af;
            af.c[0] = *(const v4h*)&E[a0];
            af.c[1] = *(const v4h*)&E[a0 + 4];
            af.c[2] = *(const v4h*)&E[a0 + 16];
            af.c[3] = *(const v4h*)&E[a0 + 20];
            acc[m] = __builtin_amdgcn_wmma_f32_16x16x32_f16(
                         false, af.v, false, bf.v, (short)0, acc[m], false, false);
        }
    }
    // relu(max(y)+b) == relu(max over valid positions + b); reduce locally,
    // then one float-atomic-max per lane into the feature slot.
    float vmax = -INFINITY;
    #pragma unroll
    for (int m = 0; m < 3; ++m) {
        #pragma unroll
        for (int r = 0; r < 8; ++r) {
            int pos = m * 16 + hs * 8 + r;           // C: VGPR r -> M=r(+8)
            float v = (pos < P) ? acc[m][r] : -INFINITY;
            vmax = fmaxf(vmax, v);
        }
    }
    atomicMaxFloatBits(&featbits[fsIdx * 64 + nt * 16 + laneM], vmax);
    __syncthreads();

    if (tid < D_CAT) {
        const int fi   = tid / 64;
        const int filt = tid % 64;
        const float b = (fi == 0) ? cb3[filt] : (fi == 1) ? cb4[filt] : cb5[filt];
        float v = __int_as_float(featbits[tid]) + b;
        feats[tid] = v > 0.0f ? v : 0.0f;
    }
    __syncthreads();

    if (tid < D_OUT) {
        float a = trans_b[tid];
        const float* w = trans_w + tid * D_CAT;
        #pragma unroll 4
        for (int c = 0; c < D_CAT; ++c) a += w[c] * feats[c];
        s_utt[n * D_OUT + tid] = tanhf(a);
    }
}

// ---------------------------------------------------------------------------
// Kernel 2: batched-window GRU via WMMA.
//   16 windows/block; per step: gi = X(16x128f16) @ wihB^T, gh = H @ whhB^T
//   (19 N-tiles x 4 K-steps each), f32 recurrent state kept in LDS.
// ---------------------------------------------------------------------------
__global__ void __launch_bounds__(256)
gru_wmma_kernel(const float* __restrict__ s_utt,
                const _Float16* __restrict__ wihB,
                const _Float16* __restrict__ whhB,
                const float* __restrict__ bih, const float* __restrict__ bhh,
                float* __restrict__ mem_bank) {
    __shared__ _Float16 XA[GW * GRU_KP];     // x as f16, K-padded
    __shared__ _Float16 HA[GW * GRU_KP];     // h as f16, K-padded
    __shared__ float x32[GW * D_OUT], h32[GW * D_OUT];
    __shared__ float giL[GW * GRU_NP], ghL[GW * GRU_NP];
    __shared__ float bihS[3 * D_OUT], bhhS[3 * D_OUT];

    const int tid  = threadIdx.x;
    const int base = blockIdx.x * GW;        // first window of this block

    for (int i = tid; i < GW * GRU_KP; i += 256) { XA[i] = (_Float16)0.0f; HA[i] = (_Float16)0.0f; }
    for (int i = tid; i < GW * D_OUT;  i += 256) h32[i] = 0.0f;
    for (int i = tid; i < 3 * D_OUT;   i += 256) { bihS[i] = bih[i]; bhhS[i] = bhh[i]; }
    __syncthreads();

    const int wave = tid >> 5, lane = tid & 31, laneM = lane & 15, hs = lane >> 4;

    for (int k = 0; k < WIND; ++k) {
        // load x for this step (zero for invalid / out-of-range windows)
        for (int i = tid; i < GW * D_OUT; i += 256) {
            int w = i / D_OUT, t = i % D_OUT;
            int bw = base + w, idx = bw - 9 + k;
            float v = (bw < N_UTTS - 1 && idx >= 0) ? s_utt[idx * D_OUT + t] : 0.0f;
            x32[w * D_OUT + t] = v;
            XA[w * GRU_KP + t] = (_Float16)v;
        }
        __syncthreads();

        // 38 WMMA tile jobs: jj<19 -> gi (X @ wih), jj>=19 -> gh (H @ whh)
        for (int jj = wave; jj < 38; jj += 8) {
            const bool isGh = jj >= 19;
            const int  nt   = isGh ? jj - 19 : jj;
            const _Float16* Abuf = isGh ? HA : XA;
            const _Float16* wB   = isGh ? whhB : wihB;
            float*          outL = isGh ? ghL : giL;

            v8f acc = {};
            #pragma unroll
            for (int ks = 0; ks < GRU_KP; ks += 32) {
                const int a0 = laneM * GRU_KP + ks + hs * 8;
                AFrag af;
                af.c[0] = *(const v4h*)&Abuf[a0];
                af.c[1] = *(const v4h*)&Abuf[a0 + 4];
                af.c[2] = *(const v4h*)&Abuf[a0 + 16];
                af.c[3] = *(const v4h*)&Abuf[a0 + 20];
                const _Float16* bp = wB + (nt * 16 + laneM) * GRU_KP + ks + hs * 8;
                BFrag bf;
                bf.c[0] = *(const v8h*)(bp);
                bf.c[1] = *(const v8h*)(bp + 16);
                acc = __builtin_amdgcn_wmma_f32_16x16x32_f16(
                          false, af.v, false, bf.v, (short)0, acc, false, false);
            }
            #pragma unroll
            for (int r = 0; r < 8; ++r)           // C: row=window, col=gate
                outL[(r + hs * 8) * GRU_NP + nt * 16 + laneM] = acc[r];
        }
        __syncthreads();

        // elementwise gate update (f32), write h (f32 + f16) and mem_bank
        for (int i = tid; i < GW * D_OUT; i += 256) {
            int w = i / D_OUT, t = i % D_OUT;
            const float* gi = giL + w * GRU_NP;
            const float* gh = ghL + w * GRU_NP;
            float r  = 1.0f / (1.0f + __expf(-((gi[t] + bihS[t]) + (gh[t] + bhhS[t]))));
            float z  = 1.0f / (1.0f + __expf(-((gi[D_OUT + t] + bihS[D_OUT + t]) +
                                               (gh[D_OUT + t] + bhhS[D_OUT + t]))));
            float nn = tanhf((gi[2 * D_OUT + t] + bihS[2 * D_OUT + t]) +
                             r * (gh[2 * D_OUT + t] + bhhS[2 * D_OUT + t]));
            float hn = (1.0f - z) * nn + z * h32[w * D_OUT + t];
            h32[w * D_OUT + t] = hn;
            HA[w * GRU_KP + t] = (_Float16)hn;
            int bw = base + w;
            if (bw < N_UTTS - 1)
                mem_bank[(bw * WIND + k) * D_OUT + t] = x32[w * D_OUT + t] + hn;
        }
        __syncthreads();
    }
}

// ---------------------------------------------------------------------------
// Kernel 3: 3-hop memory attention + log-softmax classifier
//   block b < 2047: attention for utterance b+1 ; block 2047: classify row 0
// ---------------------------------------------------------------------------
__global__ void __launch_bounds__(128)
attn_cls_kernel(const float* __restrict__ s_utt,
                const float* __restrict__ mem_bank,
                const float* __restrict__ cls_w, const float* __restrict__ cls_b,
                float* __restrict__ out) {
    __shared__ float mem[WIND * D_OUT], eps[D_OUT], attnw[WIND], lg[WIND], cl[N_CLASSES];
    const int b   = blockIdx.x;
    const int tid = threadIdx.x;
    int row;

    if (b < N_UTTS - 1) {
        row = b + 1;
        for (int i = tid; i < WIND * D_OUT; i += blockDim.x)
            mem[i] = mem_bank[b * (WIND * D_OUT) + i];
        if (tid < D_OUT) eps[tid] = s_utt[row * D_OUT + tid];
        __syncthreads();

        for (int hop = 0; hop < HOPS; ++hop) {
            if (tid < WIND) {
                float L;
                if (tid >= 9 - b) {                  // valid window slot
                    L = 0.0f;
                    for (int c = 0; c < D_OUT; ++c) L += eps[c] * mem[tid * D_OUT + c];
                } else L = NEG_INF_REF;
                lg[tid] = L;
            }
            __syncthreads();
            if (tid == 0) {
                float m = lg[0];
                for (int j2 = 1; j2 < WIND; ++j2) m = fmaxf(m, lg[j2]);
                float s = 0.0f;
                for (int j2 = 0; j2 < WIND; ++j2) { float e = __expf(lg[j2] - m); attnw[j2] = e; s += e; }
                float inv = 1.0f / s;
                for (int j2 = 0; j2 < WIND; ++j2) {
                    attnw[j2] *= inv;
                    out[PRED_ELEMS + hop * ATTN_STRIDE + b * WIND + j2] = attnw[j2];
                }
            }
            __syncthreads();
            if (tid < D_OUT) {
                float e = eps[tid];
                for (int j2 = 0; j2 < WIND; ++j2) e += attnw[j2] * mem[j2 * D_OUT + tid];
                eps[tid] = e;
            }
            __syncthreads();
        }
    } else {
        row = 0;                                     // s_cont[0] = s_utt[0]
        if (tid < D_OUT) eps[tid] = s_utt[tid];
        __syncthreads();
    }

    // classifier + log_softmax
    if (tid < N_CLASSES) {
        float L = cls_b[tid];
        const float* w = cls_w + tid * D_OUT;
        for (int c = 0; c < D_OUT; ++c) L += w[c] * eps[c];
        cl[tid] = L;
    }
    __syncthreads();
    if (tid == 0) {
        float m = cl[0];
        for (int j2 = 1; j2 < N_CLASSES; ++j2) m = fmaxf(m, cl[j2]);
        float s = 0.0f;
        for (int j2 = 0; j2 < N_CLASSES; ++j2) s += __expf(cl[j2] - m);
        float lse = m + __logf(s);
        for (int j2 = 0; j2 < N_CLASSES; ++j2) out[row * N_CLASSES + j2] = cl[j2] - lse;
    }
}

// ---------------------------------------------------------------------------
extern "C" void kernel_launch(void* const* d_in, const int* in_sizes, int n_in,
                              void* d_out, int out_size, void* d_ws, size_t ws_size,
                              hipStream_t stream) {
    const int*   sents   = (const int*)  d_in[0];
    // d_in[1] = lengths (unused by reference)
    const float* emb     = (const float*)d_in[2];
    const float* trans_w = (const float*)d_in[3];
    const float* trans_b = (const float*)d_in[4];
    const float* gru_wih = (const float*)d_in[5];
    const float* gru_whh = (const float*)d_in[6];
    const float* gru_bih = (const float*)d_in[7];
    const float* gru_bhh = (const float*)d_in[8];
    const float* cls_w   = (const float*)d_in[9];
    const float* cls_b   = (const float*)d_in[10];
    const float* conv_w3 = (const float*)d_in[11];
    const float* conv_b3 = (const float*)d_in[12];
    const float* conv_w4 = (const float*)d_in[13];
    const float* conv_b4 = (const float*)d_in[14];
    const float* conv_w5 = (const float*)d_in[15];
    const float* conv_b5 = (const float*)d_in[16];
    float* out = (float*)d_out;

    // workspace layout
    float*    s_utt    = (float*)d_ws;                               // 2048*100 f32
    float*    mem_bank = s_utt + N_UTTS * D_OUT;                     // 2047*10*100 f32
    _Float16* wf16     = (_Float16*)(mem_bank + (N_UTTS - 1) * WIND * D_OUT); // WTOT
    _Float16* wihB     = wf16 + WTOT;                                // 304*128 f16
    _Float16* whhB     = wihB + GRU_W;                               // 304*128 f16

    prep_weights_kernel<<<dim3(64, 3), 128, 0, stream>>>(conv_w3, conv_w4, conv_w5, wf16);
    prep_gru_kernel<<<(GRU_W + 255) / 256, 256, 0, stream>>>(gru_wih, gru_whh, wihB, whhB);
    conv_encoder_kernel<<<N_UTTS, 384, 0, stream>>>(sents, emb, conv_b3, conv_b4, conv_b5,
                                                    trans_w, trans_b, wf16, s_utt);
    gru_wmma_kernel<<<(N_UTTS - 1 + GW - 1) / GW, 256, 0, stream>>>(
        s_utt, wihB, whhB, gru_bih, gru_bhh, mem_bank);
    attn_cls_kernel<<<N_UTTS, 128, 0, stream>>>(s_utt, mem_bank, cls_w, cls_b, out);
}
